// ATT_87522843558434
// MI455X (gfx1250) — matrix operand order repaired and checked by reading, not compiled, not run on hardware
//
#include <hip/hip_runtime.h>
#include <hip/hip_bf16.h>

// ---------------------------------------------------------------------------
// Causal single-head attention for MI455X (gfx1250).
//   bf16 WMMA pipeline + async global->LDS K staging + TDM Q staging.
//   B=8, T=2048, C_IN=512, C_OUT=512, D=64
// ---------------------------------------------------------------------------

#define BB   8
#define TT   2048
#define CIN  512
#define CO   512
#define DD   64
#define MM   (BB*TT)          // 16384 total rows

typedef __attribute__((ext_vector_type(16))) __bf16 bf16x16;
typedef __attribute__((ext_vector_type(8)))  __bf16 bf16x8;
typedef __attribute__((ext_vector_type(8)))  float  f32x8;
typedef __attribute__((ext_vector_type(4)))  unsigned int u32x4;
typedef __attribute__((ext_vector_type(8)))  unsigned int u32x8;

#define WMMA_BF16(A,B,C) \
  __builtin_amdgcn_wmma_f32_16x16x32_bf16(false,(A),false,(B),(short)0,(C),false,false)

// Async copy of 16 bytes global -> LDS, tracked by ASYNCcnt.
#define ASYNC_CP_B128(ldsoff, gaddr) \
  asm volatile("global_load_async_to_lds_b128 %0, %1, off" \
               :: "v"(ldsoff), "v"(gaddr) : "memory")

#define WAIT_ASYNC0() asm volatile("s_wait_asynccnt 0" ::: "memory")

static __device__ __forceinline__ bf16x16 cat8(bf16x8 lo, bf16x8 hi) {
  return __builtin_shufflevector(lo, hi, 0,1,2,3,4,5,6,7,8,9,10,11,12,13,14,15);
}

static __device__ __forceinline__ f32x8 zero8() {
  f32x8 z = {0.f,0.f,0.f,0.f,0.f,0.f,0.f,0.f};
  return z;
}

// ---------------- prep: f32 -> bf16 elementwise ----------------------------
__global__ void k_cvt_bf16(const float* __restrict__ s, __bf16* __restrict__ d, int n) {
  int i = blockIdx.x * blockDim.x + threadIdx.x;
  if (i < n) d[i] = (__bf16)s[i];
}

// ---------------- prep: W[K,N] f32 -> WT[N,K] bf16 -------------------------
__global__ void k_transpose_bf16(const float* __restrict__ W, __bf16* __restrict__ WT,
                                 int K, int N) {
  int i = blockIdx.x * blockDim.x + threadIdx.x;
  if (i < K * N) {
    int k = i / N, n = i - k * N;
    WT[(size_t)n * K + k] = (__bf16)W[i];
  }
}

// ---------------- GEMM: Out(bf16) = A[M,512](bf16) @ WT[N,512]^T -----------
// One wave per 32x64 output tile; k-loop is register double-buffered so the
// next k-step's fragments are in flight while the current 8 WMMAs execute.
__global__ void k_gemm_bf16(const __bf16* __restrict__ A, const __bf16* __restrict__ WT,
                            __bf16* __restrict__ Out, int N, int storeT) {
  const int Kd   = CIN;
  const int m0   = blockIdx.x * 32;
  const int n0   = blockIdx.y * 64;
  const int lane = threadIdx.x & 31;
  const int half = lane >> 4;
  const int col  = lane & 15;

  f32x8 acc[2][4];
#pragma unroll
  for (int r = 0; r < 2; ++r)
#pragma unroll
    for (int t = 0; t < 4; ++t) acc[r][t] = zero8();

  const __bf16* arow0 = A + (size_t)(m0 + col) * Kd + half * 8;
  const __bf16* arow1 = arow0 + (size_t)16 * Kd;
  const __bf16* wp0   = WT + (size_t)(n0 + col) * Kd + half * 8;

  // prologue loads (k0 = 0)
  bf16x16 a0 = cat8(*(const bf16x8*)(arow0), *(const bf16x8*)(arow0 + 16));
  bf16x16 a1 = cat8(*(const bf16x8*)(arow1), *(const bf16x8*)(arow1 + 16));
  bf16x16 bfr[4];
#pragma unroll
  for (int t = 0; t < 4; ++t) bfr[t] = cat8(*(const bf16x8*)(wp0 + (size_t)t * 16 * Kd), *(const bf16x8*)(wp0 + (size_t)t * 16 * Kd + 16));

#pragma unroll
  for (int k0 = 0; k0 < Kd; k0 += 32) {
    bf16x16 na0, na1, nb[4];
    const int k1 = k0 + 32;
    if (k1 < Kd) {            // statically resolved after full unroll
      na0 = cat8(*(const bf16x8*)(arow0 + k1), *(const bf16x8*)(arow0 + k1 + 16));
      na1 = cat8(*(const bf16x8*)(arow1 + k1), *(const bf16x8*)(arow1 + k1 + 16));
#pragma unroll
      for (int t = 0; t < 4; ++t)
        nb[t] = cat8(*(const bf16x8*)(wp0 + k1 + (size_t)t * 16 * Kd), *(const bf16x8*)(wp0 + k1 + (size_t)t * 16 * Kd + 16));
    }
#pragma unroll
    for (int t = 0; t < 4; ++t) {
      acc[0][t] = WMMA_BF16(a0, bfr[t], acc[0][t]);
      acc[1][t] = WMMA_BF16(a1, bfr[t], acc[1][t]);
    }
    if (k1 < Kd) {
      a0 = na0; a1 = na1;
#pragma unroll
      for (int t = 0; t < 4; ++t) bfr[t] = nb[t];
    }
  }

#pragma unroll
  for (int r = 0; r < 2; ++r)
#pragma unroll
    for (int t = 0; t < 4; ++t)
#pragma unroll
      for (int j = 0; j < 8; ++j) {
        int row = m0 + 16 * r + j + 8 * half;
        int c   = n0 + 16 * t + col;
        if (storeT) Out[(size_t)c * MM + row] = (__bf16)acc[r][t][j];
        else        Out[(size_t)row * N + c]  = (__bf16)acc[r][t][j];
      }
}

// ---------------- Flash attention (causal, online softmax) -----------------
// Grid: (T/16, B). Block: 128 threads = 4 waves. Wave w owns V columns
// [128w,128w+128). Sᵀ = K·Qᵀ so the softmax row index lives in the lane and P
// converts to the A-layout per-lane. K tiles (4KB, shared by all 4 waves) are
// double-buffered in LDS via async copies; the Q tile (2KB) is staged once by
// the Tensor Data Mover.
__global__ void k_attn(const __bf16* __restrict__ Q, const __bf16* __restrict__ K,
                       const __bf16* __restrict__ VT, float* __restrict__ Out) {
  __shared__ __align__(16) __bf16 kbuf[2][32 * DD];   // 2 x 4KB K double buffer
  __shared__ __align__(16) __bf16 qsm[16 * DD];       // 2KB Q tile

  const int q0   = blockIdx.x * 16;
  const int b    = blockIdx.y;
  const int tid  = threadIdx.x;
  const int lane = tid & 31;
  const int half = lane >> 4;
  const int col  = lane & 15;      // query row (for Sᵀ) / V column (for acc)
  const int n0   = (tid >> 5) * 128;
  const float scale = 0.125f;      // 1/sqrt(64)

  const __bf16* Qb = Q + (size_t)b * TT * DD;
  const __bf16* Kc = K + (size_t)b * TT * DD;

  // Prologue A: stage K tile 0 into kbuf[0] (128 threads x 32 bytes = 4KB).
  {
    unsigned lds0 = (unsigned)(size_t)(&kbuf[0][0]) + (unsigned)tid * 16u;
    unsigned long long g = (unsigned long long)(size_t)Kc + (unsigned long long)tid * 16ull;
    ASYNC_CP_B128(lds0,         g);
    ASYNC_CP_B128(lds0 + 2048u, g + 2048ull);
  }

  // Prologue B: TDM-load the 16x64 bf16 Q tile into LDS (wave 0 issues).
  if (tid < 32) {
    unsigned long long gq = (unsigned long long)(size_t)(Qb + (size_t)q0 * DD);
    unsigned qlds = (unsigned)(size_t)(&qsm[0]);
    u32x4 g0 = { 1u,                                   // count=1 valid descriptor
                 qlds,                                 // lds_addr
                 (unsigned)(gq & 0xffffffffull),       // global_addr[31:0]
                 ((unsigned)(gq >> 32) & 0x01ffffffu) | (2u << 30) }; // addr[56:32] | type=2
    u32x8 g1 = { 1u << 16,        // data_size = 2 bytes, no multicast/flags
                 64u << 16,       // tensor_dim0 = 64 (bits 79:48 -> [31:16])
                 2048u << 16,     // tensor_dim1 = 2048 (bits 111:80 -> [31:16])
                 64u << 16,       // tile_dim0 = 64 (bits 127:112 -> [31:16])
                 16u,             // tile_dim1 = 16 (bits 143:128)
                 64u,             // tensor_dim0_stride = 64 (bits 207:160)
                 0u, 0u };        // dim1 stride unused (2D tile)
    u32x4 g2 = { 0u, 0u, 0u, 0u };
    u32x4 g3 = { 0u, 0u, 0u, 0u };
    asm volatile("tensor_load_to_lds %0, %1, %2, %3"
                 :: "s"(g0), "s"(g1), "s"(g2), "s"(g3) : "memory");
    __builtin_amdgcn_s_wait_tensorcnt(0);
  }
  __syncthreads();   // Q tile visible to all waves

  // Qᵀ B-fragments (features x qrows) from LDS, kept resident.
  bf16x16 bq0 = cat8(*(const bf16x8*)(&qsm[(size_t)col * DD + half * 8]), *(const bf16x8*)(&qsm[(size_t)col * DD + half * 8 + 16]));
  bf16x16 bq1 = cat8(*(const bf16x8*)(&qsm[(size_t)col * DD + 32 + half * 8]), *(const bf16x8*)(&qsm[(size_t)col * DD + 32 + half * 8 + 16]));

  f32x8 acc[8];
#pragma unroll
  for (int t = 0; t < 8; ++t) acc[t] = zero8();

  float m = -1e30f, l = 0.0f;
  const int qrow = q0 + col;

  for (int j0 = 0; j0 < q0 + 16; j0 += 32) {
    WAIT_ASYNC0();        // own async copies landed
    __syncthreads();      // everyone's copies landed; prev readers are done
    const int cur = (j0 >> 5) & 1;

    // Kick off the next K tile into the other buffer (overlaps compute).
    if (j0 + 32 < q0 + 16) {
      unsigned lds0 = (unsigned)(size_t)(&kbuf[cur ^ 1][0]) + (unsigned)tid * 16u;
      unsigned long long g = (unsigned long long)(size_t)(Kc + (size_t)(j0 + 32) * DD)
                           + (unsigned long long)tid * 16ull;
      ASYNC_CP_B128(lds0,         g);
      ASYNC_CP_B128(lds0 + 2048u, g + 2048ull);
    }

    // ---- issue V fragment loads early; softmax VALU hides their latency ----
    const __bf16* vp = VT + (size_t)(n0 + col) * MM + (size_t)b * TT + j0 + half * 8;
    if (j0 + 32 < q0 + 16) __builtin_prefetch(vp + 32, 0, 2);
    bf16x16 bv[8];
#pragma unroll
    for (int t = 0; t < 8; ++t) bv[t] = cat8(*(const bf16x8*)(vp + (size_t)16 * t * MM), *(const bf16x8*)(vp + (size_t)16 * t * MM + 16));

    // ---- Sᵀ = K(32 keys x 64) · Qᵀ(64 x 16 rows), A-frags from LDS ----
    const __bf16* kp0 = &kbuf[cur][(size_t)col * DD + half * 8];
    const __bf16* kp1 = kp0 + (size_t)16 * DD;

    bf16x16 a00 = cat8(*(const bf16x8*)(kp0),      *(const bf16x8*)(kp0 + 16));
    bf16x16 a01 = cat8(*(const bf16x8*)(kp0 + 32), *(const bf16x8*)(kp0 + 48));
    f32x8 st0 = WMMA_BF16(a00, bq0, zero8());
    st0       = WMMA_BF16(a01, bq1, st0);

    bf16x16 a10 = cat8(*(const bf16x8*)(kp1),      *(const bf16x8*)(kp1 + 16));
    bf16x16 a11 = cat8(*(const bf16x8*)(kp1 + 32), *(const bf16x8*)(kp1 + 48));
    f32x8 st1 = WMMA_BF16(a10, bq0, zero8());
    st1       = WMMA_BF16(a11, bq1, st1);

    // ---- scale + causal mask (key index = j0 + j + 8*half (+16)) ----
    float s[16];
#pragma unroll
    for (int j = 0; j < 8; ++j) { s[j] = st0[j] * scale; s[j + 8] = st1[j] * scale; }
    if (j0 + 31 > q0) {
#pragma unroll
      for (int j = 0; j < 8; ++j) {
        int kk = j0 + j + 8 * half;
        if (kk > qrow)      s[j]     = -1e30f;
        if (kk + 16 > qrow) s[j + 8] = -1e30f;
      }
    }

    // ---- online softmax: per-lane over 16 VGPR values + half-exchange ----
    float tmax = s[0];
#pragma unroll
    for (int e = 1; e < 16; ++e) tmax = fmaxf(tmax, s[e]);
    tmax = fmaxf(tmax, __shfl_xor(tmax, 16));
    float mnew = fmaxf(m, tmax);

    float p[16]; float rsum = 0.f;
#pragma unroll
    for (int e = 0; e < 16; ++e) { p[e] = __expf(s[e] - mnew); rsum += p[e]; }
    rsum += __shfl_xor(rsum, 16);

    float alpha = __expf(m - mnew);
    l = l * alpha + rsum;
    m = mnew;

    // ---- rescale accumulators: alpha indexed by output row j+8*half ----
#pragma unroll
    for (int j = 0; j < 8; ++j) {
      float aj = __shfl(alpha, j + 8 * half);
#pragma unroll
      for (int t = 0; t < 8; ++t) acc[t][j] *= aj;
    }

    // ---- pack P into A-layout: pure per-lane (no shuffles needed) ----
    bf16x16 pa;
#pragma unroll
    for (int e = 0; e < 16; ++e) pa[e] = (__bf16)p[e];

    // ---- acc += P(16x32) · V(32 keys x 128 cols) ----
#pragma unroll
    for (int t = 0; t < 8; ++t) acc[t] = WMMA_BF16(pa, bv[t], acc[t]);
  }

  // ---- finalize: divide by row sum, store f32 output ----
#pragma unroll
  for (int j = 0; j < 8; ++j) {
    float lj  = __shfl(l, j + 8 * half);
    float inv = 1.0f / lj;
    size_t row = (size_t)b * TT + q0 + j + 8 * half;
#pragma unroll
    for (int t = 0; t < 8; ++t)
      Out[row * CO + n0 + 16 * t + col] = acc[t][j] * inv;
  }
}

// ---------------------------------------------------------------------------
extern "C" void kernel_launch(void* const* d_in, const int* in_sizes, int n_in,
                              void* d_out, int out_size, void* d_ws, size_t ws_size,
                              hipStream_t stream) {
  const float* x  = (const float*)d_in[0];
  const float* Wq = (const float*)d_in[1];
  const float* Wk = (const float*)d_in[2];
  const float* Wv = (const float*)d_in[3];
  float* out = (float*)d_out;

  // workspace layout (bf16 elements): ~38.4 MB total
  __bf16* xb  = (__bf16*)d_ws;                       // [MM, CIN]
  __bf16* WqT = xb  + (size_t)MM * CIN;              // [DD, CIN]
  __bf16* WkT = WqT + (size_t)DD * CIN;              // [DD, CIN]
  __bf16* WvT = WkT + (size_t)DD * CIN;              // [CO, CIN]
  __bf16* Qb  = WvT + (size_t)CO * CIN;              // [MM, DD]
  __bf16* Kb  = Qb  + (size_t)MM * DD;               // [MM, DD]
  __bf16* VTb = Kb  + (size_t)MM * DD;               // [CO, MM] (V transposed)

  const int nx = MM * CIN;
  k_cvt_bf16<<<dim3((nx + 255) / 256), dim3(256), 0, stream>>>(x, xb, nx);
  k_transpose_bf16<<<dim3((CIN * DD + 255) / 256), dim3(256), 0, stream>>>(Wq, WqT, CIN, DD);
  k_transpose_bf16<<<dim3((CIN * DD + 255) / 256), dim3(256), 0, stream>>>(Wk, WkT, CIN, DD);
  k_transpose_bf16<<<dim3((CIN * CO + 255) / 256), dim3(256), 0, stream>>>(Wv, WvT, CIN, CO);

  k_gemm_bf16<<<dim3(MM / 32, DD / 64), dim3(32), 0, stream>>>(xb, WqT, Qb,  DD, 0);
  k_gemm_bf16<<<dim3(MM / 32, DD / 64), dim3(32), 0, stream>>>(xb, WkT, Kb,  DD, 0);
  k_gemm_bf16<<<dim3(MM / 32, CO / 64), dim3(32), 0, stream>>>(xb, WvT, VTb, CO, 1);

  k_attn<<<dim3(TT / 16, BB), dim3(128), 0, stream>>>(Qb, Kb, VTb, out);
}
